// GlycanGNNEncoder_8942121910543
// MI455X (gfx1250) — matrix operand order repaired and checked by reading, not compiled, not run on hardware
//
#include <hip/hip_runtime.h>
#include <stdint.h>

#define NNODES  100000
#define NEDGES  300000
#define HDIM    256
#define DDIM    128
#define NLAYERS 3
#define NGRAPHS 10000
#define VVOCAB  14
#define BN_EPS  1e-5f

typedef __attribute__((ext_vector_type(16))) __bf16 v16bf;
typedef __attribute__((ext_vector_type(8)))  float v8f;

// round-to-nearest-even f32 -> bf16 (raw ushort)
__device__ __forceinline__ unsigned short f2bf(float f) {
    union { float f; uint32_t u; } c; c.f = f;
    uint32_t r = c.u + 0x7FFFu + ((c.u >> 16) & 1u);
    return (unsigned short)(r >> 16);
}

// ---------------- utility ----------------
__global__ void k_fill_f32(float* __restrict__ p, float v, int n) {
    int i = blockIdx.x * blockDim.x + threadIdx.x;
    if (i < n) p[i] = v;
}

// deg[dst] += 1 per edge (deg pre-filled with 1.0 for self-loops)
__global__ void k_deg(const int* __restrict__ dst, float* __restrict__ deg, int e) {
    int i = blockIdx.x * blockDim.x + threadIdx.x;
    if (i < e) atomicAdd(&deg[dst[i]], 1.0f);
}

__global__ void k_dinv(float* __restrict__ deg, int n) {
    int i = blockIdx.x * blockDim.x + threadIdx.x;
    if (i < n) { float d = deg[i]; deg[i] = d > 0.0f ? rsqrtf(d) : 0.0f; }
}

// Wt[l][n][k] = W[l][k][n] as bf16   (i indexes source contiguously)
__global__ void k_conv_w(const float* __restrict__ W, unsigned short* __restrict__ Wt, int total) {
    int i = blockIdx.x * blockDim.x + threadIdx.x;
    if (i >= total) return;
    int l = i / (HDIM * HDIM); int r = i - l * (HDIM * HDIM);
    int k = r / HDIM; int n = r - k * HDIM;
    Wt[(size_t)l * HDIM * HDIM + (size_t)n * HDIM + k] = f2bf(W[i]);
}

// Pt[d][k] = projW[k][d] as bf16
__global__ void k_conv_pw(const float* __restrict__ P, unsigned short* __restrict__ Pt, int total) {
    int i = blockIdx.x * blockDim.x + threadIdx.x;
    if (i >= total) return;
    int k = i / DDIM; int d = i - k * DDIM;
    Pt[(size_t)d * HDIM + k] = f2bf(P[i]);
}

// x_bf16[n][c] = bf16(emb[ids[n]][c]); one block per node, 256 threads
__global__ void k_embed(const int* __restrict__ ids, const float* __restrict__ emb,
                        unsigned short* __restrict__ xbf) {
    int n = blockIdx.x, c = threadIdx.x;
    xbf[(size_t)n * HDIM + c] = f2bf(emb[(size_t)ids[n] * HDIM + c]);
}

// ---------------- bf16 WMMA GEMM: C[M x Nc] = A[M x K] * Bt[Nc x K]^T (+bias) ----------------
// One wave computes one 16x16 output tile; K-loop of 16x16x32 WMMAs.
// Fragment layouts follow CDNA5 ISA 7.12.2 (wave32).
__global__ void __launch_bounds__(256)
k_gemm_bf16(const unsigned short* __restrict__ A, const unsigned short* __restrict__ Bt,
            float* __restrict__ C, const float* __restrict__ bias,
            int M, int Ncols, int K) {
    int lane = threadIdx.x & 31;
    int wave = threadIdx.x >> 5;
    int gw   = blockIdx.x * 8 + wave;
    int tilesN = Ncols >> 4;
    int tilesM = M >> 4;
    int tm = gw / tilesN;
    int tn = gw - tm * tilesN;
    if (tm >= tilesM) return;

    int mloc = lane & 15;   // A row / B col / C col within tile
    int khi  = lane >> 4;   // K half selector

    const uint32_t* Arow = (const uint32_t*)(A + (size_t)(tm * 16 + mloc) * K);
    const uint32_t* Brow = (const uint32_t*)(Bt + (size_t)(tn * 16 + mloc) * K);

    v8f acc = {};
    for (int k0 = 0; k0 < K; k0 += 32) {
        union { v16bf v; uint32_t u[8]; } a, b;
        int pa = (k0 >> 1) + khi * 4;
        #pragma unroll
        for (int v = 0; v < 4; ++v) {
            a.u[v]     = Arow[pa + v];        // K = k0 + khi*8 + 2v (,+1)
            a.u[4 + v] = Arow[pa + 8 + v];    // K = k0 + khi*8 + 16 + 2v (,+1)
        }
        int pb = (k0 >> 1) + khi * 4;
        #pragma unroll
        for (int v = 0; v < 4; ++v) {
            b.u[v]     = Brow[pb + v];        // K = k0 + khi*8 + 2v (,+1)
            b.u[4 + v] = Brow[pb + 8 + v];    // K = k0 + khi*8 + 16 + 2v (,+1)
        }
        acc = __builtin_amdgcn_wmma_f32_16x16x32_bf16(
            false, a.v, false, b.v, (short)0, acc, false, false);
    }

    float bv = bias ? bias[tn * 16 + mloc] : 0.0f;
    #pragma unroll
    for (int r = 0; r < 8; ++r) {
        int m = r + khi * 8;                  // C layout: VGPR r -> M = r (+8 for hi lanes)
        C[(size_t)(tm * 16 + m) * Ncols + tn * 16 + mloc] = acc[r] + bv;
    }
}

// ---------------- edge scatter-add: agg[dst] += h[src] * dinv[src]*dinv[dst] ----------------
// one thread per (edge, 4-float chunk)
__global__ void k_edge_agg(const int* __restrict__ src, const int* __restrict__ dst,
                           const float* __restrict__ dinv, const float* __restrict__ h,
                           float* __restrict__ agg) {
    const int chunks = HDIM / 4;
    int gid = blockIdx.x * blockDim.x + threadIdx.x;
    if (gid >= NEDGES * chunks) return;
    int e = gid / chunks;
    int c = (gid - e * chunks) * 4;
    int s = src[e], d = dst[e];
    float nrm = dinv[s] * dinv[d];
    float4 v = *(const float4*)(h + (size_t)s * HDIM + c);
    float* o = agg + (size_t)d * HDIM + c;
    atomicAdd(o + 0, v.x * nrm);
    atomicAdd(o + 1, v.y * nrm);
    atomicAdd(o + 2, v.z * nrm);
    atomicAdd(o + 3, v.w * nrm);
}

// self-loop + bias + BatchNorm(eval) + ReLU; writes f32 and bf16 activations
__global__ void k_post(const float* __restrict__ agg, const float* __restrict__ h,
                       const float* __restrict__ dinv,
                       const float* __restrict__ bs, const float* __restrict__ gamma,
                       const float* __restrict__ beta, const float* __restrict__ mean,
                       const float* __restrict__ var,
                       float* __restrict__ x32, unsigned short* __restrict__ xbf) {
    int n = blockIdx.x, c = threadIdx.x;
    size_t idx = (size_t)n * HDIM + c;
    float di = dinv[n];
    float v = agg[idx] + h[idx] * di * di + bs[c];
    v = (v - mean[c]) * rsqrtf(var[c] + BN_EPS) * gamma[c] + beta[c];
    v = fmaxf(v, 0.0f);
    x32[idx] = v;
    xbf[idx] = f2bf(v);
}

// ---------------- mean pool ----------------
__global__ void k_pool_count(const int* __restrict__ bid, float* __restrict__ counts) {
    int i = blockIdx.x * blockDim.x + threadIdx.x;
    if (i < NNODES) atomicAdd(&counts[bid[i]], 1.0f);
}

__global__ void k_pool_sum(const int* __restrict__ bid, const float* __restrict__ x32,
                           float* __restrict__ sums) {
    const int chunks = HDIM / 4;
    int gid = blockIdx.x * blockDim.x + threadIdx.x;
    if (gid >= NNODES * chunks) return;
    int n = gid / chunks;
    int c = (gid - n * chunks) * 4;
    int g = bid[n];
    float4 v = *(const float4*)(x32 + (size_t)n * HDIM + c);
    float* o = sums + (size_t)g * HDIM + c;
    atomicAdd(o + 0, v.x);
    atomicAdd(o + 1, v.y);
    atomicAdd(o + 2, v.z);
    atomicAdd(o + 3, v.w);
}

__global__ void k_pool_div(const float* __restrict__ sums, const float* __restrict__ counts,
                           unsigned short* __restrict__ pooledbf) {
    int g = blockIdx.x, c = threadIdx.x;
    float cnt = fmaxf(counts[g], 1.0f);
    pooledbf[(size_t)g * HDIM + c] = f2bf(sums[(size_t)g * HDIM + c] / cnt);
}

// ---------------- launch ----------------
static inline size_t alignup(size_t v) { return (v + 255) & ~(size_t)255; }

extern "C" void kernel_launch(void* const* d_in, const int* in_sizes, int n_in,
                              void* d_out, int out_size, void* d_ws, size_t ws_size,
                              hipStream_t stream) {
    const int*   node_ids = (const int*)d_in[0];
    const int*   ei       = (const int*)d_in[1];   // [2, E]: src then dst
    const int*   batch    = (const int*)d_in[2];
    const float* emb      = (const float*)d_in[3];
    const float* Ws       = (const float*)d_in[4];
    const float* bs       = (const float*)d_in[5];
    const float* gammas   = (const float*)d_in[6];
    const float* betas    = (const float*)d_in[7];
    const float* rmean    = (const float*)d_in[8];
    const float* rvar     = (const float*)d_in[9];
    const float* projW    = (const float*)d_in[10];
    const float* projb    = (const float*)d_in[11];
    float* out = (float*)d_out;

    const int* esrc = ei;
    const int* edst = ei + NEDGES;

    // ---- workspace partition ----
    uint8_t* ws = (uint8_t*)d_ws;
    size_t off = 0;
    float* dinv = (float*)(ws + off);          off = alignup(off + (size_t)NNODES * 4);
    float* h32  = (float*)(ws + off);          off = alignup(off + (size_t)NNODES * HDIM * 4);
    float* agg  = (float*)(ws + off);          off = alignup(off + (size_t)NNODES * HDIM * 4);
    float* x32  = (float*)(ws + off);          off = alignup(off + (size_t)NNODES * HDIM * 4);
    unsigned short* xbf = (unsigned short*)(ws + off); off = alignup(off + (size_t)NNODES * HDIM * 2);
    unsigned short* wt  = (unsigned short*)(ws + off); off = alignup(off + (size_t)NLAYERS * HDIM * HDIM * 2);
    unsigned short* pwt = (unsigned short*)(ws + off); off = alignup(off + (size_t)HDIM * DDIM * 2);
    float* sums   = (float*)(ws + off);        off = alignup(off + (size_t)NGRAPHS * HDIM * 4);
    float* counts = (float*)(ws + off);        off = alignup(off + (size_t)NGRAPHS * 4);
    unsigned short* pbf = (unsigned short*)(ws + off); off = alignup(off + (size_t)NGRAPHS * HDIM * 2);

    const int T = 256;
    auto blocks = [](long n, int t) { return (int)((n + t - 1) / t); };

    // ---- degree / normalization ----
    k_fill_f32<<<blocks(NNODES, T), T, 0, stream>>>(dinv, 1.0f, NNODES);
    k_deg<<<blocks(NEDGES, T), T, 0, stream>>>(edst, dinv, NEDGES);
    k_dinv<<<blocks(NNODES, T), T, 0, stream>>>(dinv, NNODES);

    // ---- weight conversion (bf16, transposed to [n][k]) ----
    k_conv_w<<<blocks((long)NLAYERS * HDIM * HDIM, T), T, 0, stream>>>(Ws, wt, NLAYERS * HDIM * HDIM);
    k_conv_pw<<<blocks((long)HDIM * DDIM, T), T, 0, stream>>>(projW, pwt, HDIM * DDIM);

    // ---- embedding gather ----
    k_embed<<<NNODES, HDIM, 0, stream>>>(node_ids, emb, xbf);

    // ---- GCN layers ----
    for (int l = 0; l < NLAYERS; ++l) {
        // h = x @ W[l]  (bf16 WMMA, f32 accumulate)
        int waves = (NNODES / 16) * (HDIM / 16);
        k_gemm_bf16<<<blocks(waves, 8), T, 0, stream>>>(xbf, wt + (size_t)l * HDIM * HDIM,
                                                        h32, nullptr, NNODES, HDIM, HDIM);
        // agg = 0; scatter-add messages over edges
        k_fill_f32<<<blocks((long)NNODES * HDIM, T), T, 0, stream>>>(agg, 0.0f, NNODES * HDIM);
        k_edge_agg<<<blocks((long)NEDGES * (HDIM / 4), T), T, 0, stream>>>(esrc, edst, dinv, h32, agg);
        // self-loop + bias + BN + ReLU
        k_post<<<NNODES, HDIM, 0, stream>>>(agg, h32, dinv,
                                            bs + l * HDIM, gammas + l * HDIM, betas + l * HDIM,
                                            rmean + l * HDIM, rvar + l * HDIM, x32, xbf);
    }

    // ---- global mean pool ----
    k_fill_f32<<<blocks((long)NGRAPHS * HDIM, T), T, 0, stream>>>(sums, 0.0f, NGRAPHS * HDIM);
    k_fill_f32<<<blocks(NGRAPHS, T), T, 0, stream>>>(counts, 0.0f, NGRAPHS);
    k_pool_count<<<blocks(NNODES, T), T, 0, stream>>>(batch, counts);
    k_pool_sum<<<blocks((long)NNODES * (HDIM / 4), T), T, 0, stream>>>(batch, x32, sums);
    k_pool_div<<<NGRAPHS, HDIM, 0, stream>>>(sums, counts, pbf);

    // ---- final projection: out = pooled @ projW + projb ----
    int fwaves = (NGRAPHS / 16) * (DDIM / 16);
    k_gemm_bf16<<<blocks(fwaves, 8), T, 0, stream>>>(pbf, pwt, out, projb, NGRAPHS, DDIM, HDIM);

    (void)in_sizes; (void)n_in; (void)out_size; (void)ws_size;
}
